// CelestialGraphCombiner_40638980554829
// MI455X (gfx1250) — matrix operand order repaired and checked by reading, not compiled, not run on hardware
//
#include <hip/hip_runtime.h>
#include <hip/hip_bf16.h>

// ---------------------------------------------------------------------------
// Types for CDNA5 WMMA (wave32, 16x16x32 f16 -> f32)
// ---------------------------------------------------------------------------
typedef __attribute__((ext_vector_type(16))) _Float16 v16h;
typedef __attribute__((ext_vector_type(4)))  _Float16 v4h;
typedef __attribute__((ext_vector_type(8)))  float    v8f;

union V16U { uint4 u4[2]; v16h v; };

__device__ inline v8f wmma_f32(v16h a, v16h b, v8f c) {
  return __builtin_amdgcn_wmma_f32_16x16x32_f16(
      /*neg_a=*/false, a, /*neg_b=*/false, b,
      /*c_mod=*/(short)0, c, /*reuse_a=*/false, /*reuse_b=*/false);
}

__device__ inline v4h cvt4(float4 f) {
  return (v4h){(_Float16)f.x, (_Float16)f.y, (_Float16)f.z, (_Float16)f.w};
}

// A-matrix fragment (16x32, f16). LDS layout: As[row][k], row stride 32 halves.
// Lane L holds row M=L%16. Halves h<8 -> k = h + (laneHi?8:0);
// halves h>=8 -> k = h+8 + (laneHi?8:0).  => two contiguous 16B chunks.
__device__ inline v16h ldsA_frag(const _Float16* As, int r0, int laneHi, int lane16) {
  const _Float16* p = As + (r0 + lane16) * 32 + laneHi * 8;
  V16U f;
  f.u4[0] = *(const uint4*)(p);
  f.u4[1] = *(const uint4*)(p + 16);
  return f.v;
}

// B-matrix fragment (32x16, f16) from transposed LDS: BTs[col][k], stride 32.
// Lane L holds col N=L%16. Halves h<8 -> k = h + (laneHi?8:0);
// halves h>=8 -> k = h+8 + (laneHi?8:0).  => two contiguous 16B chunks.
__device__ inline v16h ldsBT_frag(const _Float16* BTs, int c0, int laneHi, int lane16) {
  const _Float16* p = BTs + (c0 + lane16) * 32 + laneHi * 8;
  V16U f;
  f.u4[0] = *(const uint4*)(p);
  f.u4[1] = *(const uint4*)(p + 16);
  return f.v;
}

__device__ inline float gelu_f(float x) {
  return 0.5f * x * (1.0f + erff(x * 0.70710678118654752f));
}
__device__ inline float sigm_f(float x) { return 1.0f / (1.0f + __expf(-x)); }

// ---------------------------------------------------------------------------
// Generic WMMA GEMM: C[M,N] = act(A[M,K] @ B[K,N] + bias)
// Tile 64x64, BK=32, 256 threads (8 waves), each wave -> 16x32 of output.
// Vectorized LDS staging: per k-tile each thread does 2 b128 global loads for A
// and 2 for B (all issued before any wait), f32->f16 convert, b64/b16 LDS store.
// Optional per-batch expert weights: blockIdx.z = batch, B += regime[z]*estride.
// M % 64 == 0, N % 64 == 0, K % 32 == 0 (all shapes here satisfy this).
// ---------------------------------------------------------------------------
__global__ __launch_bounds__(256) void gemm_kernel(
    const float* __restrict__ A, const float* __restrict__ Bw,
    const float* __restrict__ bias, float* __restrict__ C,
    int M, int N, int K, int act,
    const int* regime, int estride, int rpb) {
  __shared__ alignas(64) _Float16 As[64 * 32];
  __shared__ alignas(64) _Float16 BTs[64 * 32];
  int bz = blockIdx.z;
  if (regime) {
    int r = regime[bz];
    Bw += r * estride;
    if (bias) bias += r * N;
    A += bz * rpb * K;
    C += bz * rpb * N;
  }
  int row0 = blockIdx.y * 64, col0 = blockIdx.x * 64;
  int tid = threadIdx.x;
  int lane = tid & 31, wave = tid >> 5;
  int lane16 = lane & 15, laneHi = lane >> 4;
  int wrow = (wave & 3) * 16, wcol = (wave >> 2) * 32;
  // per-thread staging coordinates (compile-time trip counts)
  int ar0 = tid >> 3, ac0 = (tid & 7) << 2;       // A chunk 0: row, 4*k
  int ar1 = (tid + 256) >> 3, ac1 = ac0;          // A chunk 1
  int bk0i = tid >> 4, bc0 = (tid & 15) << 2;     // B chunk 0: k, 4*col
  int bk1i = (tid + 256) >> 4, bc1 = bc0;         // B chunk 1
  v8f acc0 = {}, acc1 = {};
  for (int k0 = 0; k0 < K; k0 += 32) {
    float4 fa0 = *(const float4*)(A + (row0 + ar0) * K + k0 + ac0);
    float4 fa1 = *(const float4*)(A + (row0 + ar1) * K + k0 + ac1);
    float4 fb0 = *(const float4*)(Bw + (k0 + bk0i) * N + col0 + bc0);
    float4 fb1 = *(const float4*)(Bw + (k0 + bk1i) * N + col0 + bc1);
    if (k0 + 32 < K) {  // prefetch next k-tile (global_prefetch_b8)
      __builtin_prefetch(A + (row0 + ar0) * K + k0 + 32 + ac0, 0, 3);
      __builtin_prefetch(Bw + (k0 + 32 + bk0i) * N + col0 + bc0, 0, 3);
    }
    *(v4h*)(As + ar0 * 32 + ac0) = cvt4(fa0);
    *(v4h*)(As + ar1 * 32 + ac1) = cvt4(fa1);
    BTs[(bc0 + 0) * 32 + bk0i] = (_Float16)fb0.x;
    BTs[(bc0 + 1) * 32 + bk0i] = (_Float16)fb0.y;
    BTs[(bc0 + 2) * 32 + bk0i] = (_Float16)fb0.z;
    BTs[(bc0 + 3) * 32 + bk0i] = (_Float16)fb0.w;
    BTs[(bc1 + 0) * 32 + bk1i] = (_Float16)fb1.x;
    BTs[(bc1 + 1) * 32 + bk1i] = (_Float16)fb1.y;
    BTs[(bc1 + 2) * 32 + bk1i] = (_Float16)fb1.z;
    BTs[(bc1 + 3) * 32 + bk1i] = (_Float16)fb1.w;
    __syncthreads();
    v16h a  = ldsA_frag(As, wrow, laneHi, lane16);
    v16h b0 = ldsBT_frag(BTs, wcol, laneHi, lane16);
    v16h b1 = ldsBT_frag(BTs, wcol + 16, laneHi, lane16);
    acc0 = wmma_f32(a, b0, acc0);
    acc1 = wmma_f32(a, b1, acc1);
    __syncthreads();
  }
  int mbase = row0 + wrow + laneHi * 8;
  int n0 = col0 + wcol + lane16;
  int n1 = n0 + 16;
  float bb0 = bias ? bias[n0] : 0.f;
  float bb1 = bias ? bias[n1] : 0.f;
#pragma unroll
  for (int r = 0; r < 8; ++r) {
    int m = mbase + r;
    float va = acc0[r] + bb0, vb = acc1[r] + bb1;
    if (act == 1) { va = gelu_f(va); vb = gelu_f(vb); }
    C[m * N + n0] = va;
    C[m * N + n1] = vb;
  }
}

// ---------------------------------------------------------------------------
// Flash attention, one wave per (b, h, 16-query tile). dh=16, D=128, S=1728.
// ---------------------------------------------------------------------------
__global__ __launch_bounds__(32) void attn_kernel(
    const float* __restrict__ Q, const float* __restrict__ K,
    const float* __restrict__ V, float* __restrict__ O) {
  const int S = 1728, D = 128;
  int q0 = blockIdx.x * 16, h = blockIdx.y, b = blockIdx.z;
  __shared__ alignas(64) _Float16 Qs[16 * 32];   // [q][dh pad 32]
  __shared__ alignas(64) _Float16 KsT[32 * 32];  // [key][dh pad 32]
  __shared__ alignas(64) _Float16 Ps[16 * 32];   // [q][key]
  __shared__ alignas(64) _Float16 VsT[16 * 32];  // [dh][key]
  int lane = threadIdx.x;
  int lane16 = lane & 15, laneHi = lane >> 4;
  for (int i = lane; i < 32 * 32; i += 32) KsT[i] = (_Float16)0.f;
  for (int i = lane; i < 16 * 32; i += 32) Qs[i] = (_Float16)0.f;
  asm volatile("s_wait_dscnt 0" ::: "memory");
  if (lane < 16) {
    const float4* qp = (const float4*)(Q + (b * S + q0 + lane) * D + h * 16);
    float4 q0v = qp[0], q1v = qp[1], q2v = qp[2], q3v = qp[3];
    *(v4h*)(Qs + lane * 32 + 0)  = cvt4(q0v);
    *(v4h*)(Qs + lane * 32 + 4)  = cvt4(q1v);
    *(v4h*)(Qs + lane * 32 + 8)  = cvt4(q2v);
    *(v4h*)(Qs + lane * 32 + 12) = cvt4(q3v);
  }
  asm volatile("s_wait_dscnt 0" ::: "memory");
  v16h aQ = ldsA_frag(Qs, 0, laneHi, lane16);
  v8f Oacc = {};
  float m_i[8], l_i[8];
#pragma unroll
  for (int r = 0; r < 8; ++r) { m_i[r] = -1e30f; l_i[r] = 0.f; }

  for (int kc = 0; kc < S; kc += 32) {
    const float4* kp = (const float4*)(K + (b * S + kc + lane) * D + h * 16);
    const float4* vp = (const float4*)(V + (b * S + kc + lane) * D + h * 16);
    float4 k0v = kp[0], k1v = kp[1], k2v = kp[2], k3v = kp[3];
    float4 v0v = vp[0], v1v = vp[1], v2v = vp[2], v3v = vp[3];
    if (kc + 32 < S) {  // prefetch next chunk (global_prefetch_b8)
      __builtin_prefetch(K + (b * S + kc + 32 + lane) * D + h * 16, 0, 3);
      __builtin_prefetch(V + (b * S + kc + 32 + lane) * D + h * 16, 0, 3);
    }
    *(v4h*)(KsT + lane * 32 + 0)  = cvt4(k0v);
    *(v4h*)(KsT + lane * 32 + 4)  = cvt4(k1v);
    *(v4h*)(KsT + lane * 32 + 8)  = cvt4(k2v);
    *(v4h*)(KsT + lane * 32 + 12) = cvt4(k3v);
    VsT[0  * 32 + lane] = (_Float16)v0v.x;
    VsT[1  * 32 + lane] = (_Float16)v0v.y;
    VsT[2  * 32 + lane] = (_Float16)v0v.z;
    VsT[3  * 32 + lane] = (_Float16)v0v.w;
    VsT[4  * 32 + lane] = (_Float16)v1v.x;
    VsT[5  * 32 + lane] = (_Float16)v1v.y;
    VsT[6  * 32 + lane] = (_Float16)v1v.z;
    VsT[7  * 32 + lane] = (_Float16)v1v.w;
    VsT[8  * 32 + lane] = (_Float16)v2v.x;
    VsT[9  * 32 + lane] = (_Float16)v2v.y;
    VsT[10 * 32 + lane] = (_Float16)v2v.z;
    VsT[11 * 32 + lane] = (_Float16)v2v.w;
    VsT[12 * 32 + lane] = (_Float16)v3v.x;
    VsT[13 * 32 + lane] = (_Float16)v3v.y;
    VsT[14 * 32 + lane] = (_Float16)v3v.z;
    VsT[15 * 32 + lane] = (_Float16)v3v.w;
    asm volatile("s_wait_dscnt 0" ::: "memory");
    v16h bk0 = ldsBT_frag(KsT, 0, laneHi, lane16);
    v16h bk1 = ldsBT_frag(KsT, 16, laneHi, lane16);
    v8f z = {};
    v8f s0 = wmma_f32(aQ, bk0, z);
    v8f s1 = wmma_f32(aQ, bk1, z);
#pragma unroll
    for (int r = 0; r < 8; ++r) {
      float x0 = s0[r] * 0.25f, x1 = s1[r] * 0.25f;  // 1/sqrt(dh=16)
      float cm = fmaxf(x0, x1);
      cm = fmaxf(cm, __shfl_xor(cm, 1, 32));
      cm = fmaxf(cm, __shfl_xor(cm, 2, 32));
      cm = fmaxf(cm, __shfl_xor(cm, 4, 32));
      cm = fmaxf(cm, __shfl_xor(cm, 8, 32));
      float mnew = fmaxf(m_i[r], cm);
      float corr = __expf(m_i[r] - mnew);
      float p0 = __expf(x0 - mnew), p1 = __expf(x1 - mnew);
      float ps = p0 + p1;
      ps += __shfl_xor(ps, 1, 32);
      ps += __shfl_xor(ps, 2, 32);
      ps += __shfl_xor(ps, 4, 32);
      ps += __shfl_xor(ps, 8, 32);
      l_i[r] = l_i[r] * corr + ps;
      m_i[r] = mnew;
      Oacc[r] *= corr;
      int mrow = r + laneHi * 8;
      Ps[mrow * 32 + lane16] = (_Float16)p0;
      Ps[mrow * 32 + 16 + lane16] = (_Float16)p1;
    }
    asm volatile("s_wait_dscnt 0" ::: "memory");
    v16h aP = ldsA_frag(Ps, 0, laneHi, lane16);
    v16h bV = ldsBT_frag(VsT, 0, laneHi, lane16);
    Oacc = wmma_f32(aP, bV, Oacc);
  }
#pragma unroll
  for (int r = 0; r < 8; ++r) {
    int mrow = r + laneHi * 8;
    O[(b * S + q0 + mrow) * D + h * 16 + lane16] = Oacc[r] / l_i[r];
  }
}

// ---------------------------------------------------------------------------
// Small VALU kernels
// ---------------------------------------------------------------------------
struct Lin3P { const float *b0, *w0, *b1, *w1, *b2, *w2; };
struct EdgeP { const float *lnb, *lng; Lin3P l; };
struct EdgeAll { EdgeP e[3]; const float* emb; };
struct InterNets { Lin3P net[3]; };

__device__ inline float block_sum128(float v, float* red, int tid) {
  red[tid] = v; __syncthreads();
  for (int s = 64; s > 0; s >>= 1) { if (tid < s) red[tid] += red[tid + s]; __syncthreads(); }
  float r = red[0]; __syncthreads();
  return r;
}

// edge value -> MLP(1->32->64->128) -> LN -> + type emb -> X[b, t*576+e, :]
__global__ __launch_bounds__(128) void edge_tf_kernel(
    const float* astro, const float* learned, const float* attn,
    EdgeAll P, float* X) {
  int e = blockIdx.x, t = blockIdx.y, b = blockIdx.z;
  int tid = threadIdx.x;
  __shared__ float h1[32], h2[64], red[128];
  const float* Es = (t == 0) ? astro : ((t == 1) ? learned : attn);
  float Ev = Es[b * 576 + e];
  EdgeP p = P.e[t];
  if (tid < 32) h1[tid] = gelu_f(Ev * p.l.w0[tid] + p.l.b0[tid]);
  __syncthreads();
  if (tid < 64) {
    float s = p.l.b1[tid];
    for (int k = 0; k < 32; ++k) s += h1[k] * p.l.w1[k * 64 + tid];
    h2[tid] = gelu_f(s);
  }
  __syncthreads();
  float o = p.l.b2[tid];
  for (int k = 0; k < 64; ++k) o += h2[k] * p.l.w2[k * 128 + tid];
  float mean = block_sum128(o, red, tid) * (1.f / 128.f);
  float d = o - mean;
  float var = block_sum128(d * d, red, tid) * (1.f / 128.f);
  float y = d * rsqrtf(var + 1e-5f) * p.lng[tid] + p.lnb[tid] + P.emb[t * 128 + tid];
  X[(b * 1728 + t * 576 + e) * 128 + tid] = y;
}

// x = LN(x + res); bcast_rows!=0 -> res broadcast per batch of bcast_rows rows
__global__ __launch_bounds__(128) void addln_kernel(
    float* x, const float* res, const float* g, const float* bta, int bcast_rows) {
  int row = blockIdx.x, j = threadIdx.x;
  __shared__ float red[128];
  int xo = row * 128 + j;
  float r = bcast_rows ? res[(row / bcast_rows) * 128 + j] : res[xo];
  float v = x[xo] + r;
  float mean = block_sum128(v, red, j) * (1.f / 128.f);
  float d = v - mean;
  float var = block_sum128(d * d, red, j) * (1.f / 128.f);
  x[xo] = d * rsqrtf(var + 1e-5f) * g[j] + bta[j];
}

// regime LN: out = LN(h2) * g[r] + bt[r]
__global__ __launch_bounds__(128) void regime_ln_kernel(
    const float* h2, float* outc, const int* regime, const float* g4, const float* b4) {
  int row = blockIdx.x, j = threadIdx.x;
  int b = row / 576, r = regime[b];
  __shared__ float red[128];
  float v = h2[row * 128 + j];
  float mean = block_sum128(v, red, j) * (1.f / 128.f);
  float d = v - mean;
  float var = block_sum128(d * d, red, j) * (1.f / 128.f);
  outc[row * 128 + j] = d * rsqrtf(var + 1e-5f) * g4[r * 128 + j] + b4[r * 128 + j];
}

// ca[b] = (ctx@w1 + b1) @ w2 + b2
__global__ __launch_bounds__(128) void ctx_mlp2_kernel(
    const float* ctx, const float* w1, const float* b1,
    const float* w2, const float* b2, float* out) {
  __shared__ float t1[128];
  int b = blockIdx.x, j = threadIdx.x;
  float s = b1[j];
  for (int k = 0; k < 128; ++k) s += ctx[b * 128 + k] * w1[k * 128 + j];
  t1[j] = s; __syncthreads();
  float o = b2[j];
  for (int k = 0; k < 128; ++k) o += t1[k] * w2[k * 128 + j];
  out[b * 128 + j] = o;
}

// gate[b] = sigmoid(ctx@w + b)
__global__ __launch_bounds__(128) void ctx_gate_kernel(
    const float* ctx, const float* w, const float* bb, float* out) {
  int b = blockIdx.x, j = threadIdx.x;
  float s = bb[j];
  for (int k = 0; k < 128; ++k) s += ctx[b * 128 + k] * w[k * 128 + j];
  out[b * 128 + j] = sigm_f(s);
}

// fused = softmax(comb_w) . x streams
__global__ void combine_kernel(const float* x, const float* cw, float* fused) {
  int i = blockIdx.x * 256 + threadIdx.x;  // over 4*576*128
  int d = i & 127, s = (i >> 7) % 576, b = i / (576 * 128);
  float c0 = cw[0], c1 = cw[1], c2 = cw[2];
  float mx = fmaxf(c0, fmaxf(c1, c2));
  float e0 = __expf(c0 - mx), e1 = __expf(c1 - mx), e2 = __expf(c2 - mx);
  float inv = 1.f / (e0 + e1 + e2);
  const float* xb = x + (b * 1728 + s) * 128 + d;
  const int ts = 576 * 128;
  fused[i] = (e0 * xb[0] + e1 * xb[ts] + e2 * xb[2 * ts]) * inv;
}

// combined += (sum of 3 gated pair-MLPs)/3
__global__ __launch_bounds__(128) void inter_kernel(
    const float* astro, const float* learned, const float* attn,
    InterNets P, const float* gates, float* comb) {
  int row = blockIdx.x;
  int b = row / 576;
  int tid = threadIdx.x;
  __shared__ float h1[32], h2[64];
  float a = astro[row], le = learned[row], at = attn[row];
  float acc = 0.f;
  for (int pi = 0; pi < 3; ++pi) {
    float in0 = (pi == 2) ? le : a;
    float in1 = (pi == 1) ? le : at;
    Lin3P n = P.net[pi];
    __syncthreads();
    if (tid < 32) h1[tid] = gelu_f(in0 * n.w0[tid] + in1 * n.w0[32 + tid] + n.b0[tid]);
    __syncthreads();
    if (tid < 64) {
      float s = n.b1[tid];
      for (int k = 0; k < 32; ++k) s += h1[k] * n.w1[k * 64 + tid];
      h2[tid] = gelu_f(s);
    }
    __syncthreads();
    float o = n.b2[tid];
    for (int k = 0; k < 64; ++k) o += h2[k] * n.w2[k * 128 + tid];
    acc += o * gates[(pi * 4 + b) * 128 + tid];
  }
  comb[row * 128 + tid] += acc * (1.f / 3.f);
}

// cc[row] = [combined(128) | ctx(128)]
__global__ void concat_kernel(const float* comb, const float* ctx, float* cc) {
  int i = blockIdx.x * 256 + threadIdx.x;  // 2304*256
  int j = i & 255, row = i >> 8, b = row / 576;
  cc[i] = (j < 128) ? comb[row * 128 + j] : ctx[b * 128 + (j - 128)];
}

// out = tanh(gelu((comb*sigmoid(c3)) @ w0 + b0) @ w1 + b1)
__global__ __launch_bounds__(128) void final_kernel(
    const float* comb, const float* c3, const float* w0, const float* b0,
    const float* w1, const float* b1, float* out) {
  int row = blockIdx.x, j = threadIdx.x;
  __shared__ float y[128], red[128];
  y[j] = comb[row * 128 + j] * sigm_f(c3[row * 128 + j]);
  __syncthreads();
  float h = b0[j];
  for (int k = 0; k < 128; ++k) h += y[k] * w0[k * 128 + j];
  h = gelu_f(h);
  float s = block_sum128(h * w1[j], red, j);
  if (j == 0) out[row] = tanhf(s + b1[0]);
}

// ---------------------------------------------------------------------------
// Host launcher. Param indices follow jax tree flatten (sorted dict keys).
// ---------------------------------------------------------------------------
extern "C" void kernel_launch(void* const* d_in, const int* in_sizes, int n_in,
                              void* d_out, int out_size, void* d_ws, size_t ws_size,
                              hipStream_t stream) {
  (void)in_sizes; (void)n_in; (void)out_size; (void)ws_size;
  auto F = [&](int i) { return (const float*)d_in[i]; };
  const float* ctx = F(3);
  const int* regime = (const int*)d_in[4];
  const int S = 1728, ROWS = 4 * S;  // 6912

  float* ws = (float*)d_ws;
  size_t off = 0;
  auto alloc = [&](size_t n) { float* p = ws + off; off += n; return p; };
  float* X  = alloc((size_t)ROWS * 128);
  float* Qb = alloc((size_t)ROWS * 128);
  float* Kb = alloc((size_t)ROWS * 128);
  float* Vb = alloc((size_t)ROWS * 128);
  float* Tb = alloc((size_t)ROWS * 128);
  float* H1 = alloc((size_t)ROWS * 512);
  float* CA = alloc(2 * 4 * 128);
  float* GT = alloc(3 * 4 * 128);

  // interaction gates: param groups astro_attention(89), astro_learned(97), learned_attention(105)
  ctx_gate_kernel<<<4, 128, 0, stream>>>(ctx, F(90), F(89), GT + 0 * 512);
  ctx_gate_kernel<<<4, 128, 0, stream>>>(ctx, F(98), F(97), GT + 1 * 512);
  ctx_gate_kernel<<<4, 128, 0, stream>>>(ctx, F(106), F(105), GT + 2 * 512);
  // cross-attn collapse per layer: ca = (ctx@wv+bv)@wo+bo
  for (int l = 0; l < 2; ++l) {
    int bs = 37 + 26 * l;  // cross: bk,bo,bq,bv,wk,wo,wq,wv = +0..+7
    ctx_mlp2_kernel<<<4, 128, 0, stream>>>(ctx, F(bs + 7), F(bs + 3), F(bs + 5), F(bs + 1), CA + l * 512);
  }
  // edge transforms (x stream order: astro, learned, attention)
  EdgeAll EA;
  const int ebase[3] = {12, 28, 20};  // astro, learned, attention param groups
  for (int t = 0; t < 3; ++t) {
    int b = ebase[t];
    EA.e[t].lnb = F(b + 0); EA.e[t].lng = F(b + 1);
    EA.e[t].l = { F(b + 2), F(b + 3), F(b + 4), F(b + 5), F(b + 6), F(b + 7) };
  }
  EA.emb = F(36);
  edge_tf_kernel<<<dim3(576, 3, 4), 128, 0, stream>>>(F(0), F(1), F(2), EA, X);

  // fusion layers
  for (int l = 0; l < 2; ++l) {
    int bs = 37 + 26 * l;
    // self: bk=+18 bo=+19 bq=+20 bv=+21 wk=+22 wo=+23 wq=+24 wv=+25
    gemm_kernel<<<dim3(2, 108, 1), 256, 0, stream>>>(X, F(bs + 24), F(bs + 20), Qb, ROWS, 128, 128, 0, nullptr, 0, 0);
    gemm_kernel<<<dim3(2, 108, 1), 256, 0, stream>>>(X, F(bs + 22), F(bs + 18), Kb, ROWS, 128, 128, 0, nullptr, 0, 0);
    gemm_kernel<<<dim3(2, 108, 1), 256, 0, stream>>>(X, F(bs + 25), F(bs + 21), Vb, ROWS, 128, 128, 0, nullptr, 0, 0);
    attn_kernel<<<dim3(108, 8, 4), 32, 0, stream>>>(Qb, Kb, Vb, Tb);
    gemm_kernel<<<dim3(2, 108, 1), 256, 0, stream>>>(Tb, F(bs + 23), F(bs + 19), Qb, ROWS, 128, 128, 0, nullptr, 0, 0);
    addln_kernel<<<ROWS, 128, 0, stream>>>(X, Qb, F(bs + 13), F(bs + 12), 0);           // ln1
    addln_kernel<<<ROWS, 128, 0, stream>>>(X, CA + l * 512, F(bs + 15), F(bs + 14), S); // ln2 (+broadcast ca)
    gemm_kernel<<<dim3(8, 108, 1), 256, 0, stream>>>(X, F(bs + 9), F(bs + 8), H1, ROWS, 512, 128, 1, nullptr, 0, 0);
    gemm_kernel<<<dim3(2, 108, 1), 256, 0, stream>>>(H1, F(bs + 11), F(bs + 10), Qb, ROWS, 128, 512, 0, nullptr, 0, 0);
    addln_kernel<<<ROWS, 128, 0, stream>>>(X, Qb, F(bs + 17), F(bs + 16), 0);           // ln3
  }

  // tail-phase buffer reuse
  float* FUSED = Qb; float* R1 = Kb; float* R2 = Vb; float* COMB = Tb;
  float* CC = H1; float* C1 = H1 + 2304 * 256; float* C2 = C1 + 2304 * 128; float* C3 = C2 + 2304 * 128;

  combine_kernel<<<(4 * 576 * 128) / 256, 256, 0, stream>>>(X, F(11), FUSED);
  // regime adapter: w1=121 b1=117 w2=122 b2=118 g=120 bt=119
  gemm_kernel<<<dim3(2, 9, 4), 256, 0, stream>>>(FUSED, F(121), F(117), R1, 576, 128, 128, 1, regime, 128 * 128, 576);
  gemm_kernel<<<dim3(2, 9, 4), 256, 0, stream>>>(R1, F(122), F(118), R2, 576, 128, 128, 0, regime, 128 * 128, 576);
  regime_ln_kernel<<<2304, 128, 0, stream>>>(R2, COMB, regime, F(120), F(119));
  // pairwise interactions added into COMB
  InterNets IN;
  const int ibase[3] = {89, 97, 105};
  for (int p = 0; p < 3; ++p) {
    int b = ibase[p];
    IN.net[p] = { F(b + 2), F(b + 3), F(b + 4), F(b + 5), F(b + 6), F(b + 7) };
  }
  inter_kernel<<<2304, 128, 0, stream>>>(F(0), F(1), F(2), IN, GT, COMB);
  // calibration: calib0 b=5 w=6 (K=256), calib1 b=7 w=8, calib2 b=9 w=10
  concat_kernel<<<(2304 * 256) / 256, 256, 0, stream>>>(COMB, ctx, CC);
  gemm_kernel<<<dim3(2, 36, 1), 256, 0, stream>>>(CC, F(6), F(5), C1, 2304, 128, 256, 1, nullptr, 0, 0);
  gemm_kernel<<<dim3(2, 36, 1), 256, 0, stream>>>(C1, F(8), F(7), C2, 2304, 128, 128, 1, nullptr, 0, 0);
  gemm_kernel<<<dim3(2, 36, 1), 256, 0, stream>>>(C2, F(10), F(9), C3, 2304, 128, 128, 0, nullptr, 0, 0);
  // output head: op0.b=113 op0.w=114 op1.b=115 op1.w=116
  final_kernel<<<2304, 128, 0, stream>>>(COMB, C3, F(114), F(113), F(116), F(115), (float*)d_out);
}
